// TFGridNet_KVfusion_74466142978177
// MI455X (gfx1250) — matrix operand order repaired and checked by reading, not compiled, not run on hardware
//
#include <hip/hip_runtime.h>
#include <hip/hip_bf16.h>

// ---------------- problem constants ----------------
#define B_    4
#define C_    128
#define T_    1000
#define F_    65
#define H_    4
#define E_    8
#define DH_   32
#define TP_   199          // pooled length per side
#define TC_   398          // 2*TP_
#define EF_   520          // E*F
#define DHF_  2080         // Dh*F
#define DPAD_ 544          // EF padded to mult of 32
#define TCPAD_ 416         // Tc padded to mult of 32
#define TPAD_ 1008         // T padded to mult of 16
#define MT_   63           // TPAD/16
#define NTS_  26           // TCPAD/16
#define NTO_  130          // DHF/16
#define KTS_  17           // DPAD/32
#define KTO_  13           // TCPAD/32
#define EPS_  1e-5f

typedef __attribute__((ext_vector_type(16))) __bf16 v16bf;
typedef __attribute__((ext_vector_type(8)))  __bf16 v8bf;
typedef __attribute__((ext_vector_type(8)))  float  v8f;

// ---------------- K1: avg-pool over T + sign, build cond [B,C,Tc,F] --------
__global__ __launch_bounds__(256)
void pool_kernel(const float* __restrict__ pos, const float* __restrict__ neg,
                 float* __restrict__ cond) {
  int idx = blockIdx.x * 256 + threadIdx.x;
  const int total = B_ * C_ * TC_ * F_;
  if (idx >= total) return;
  int f  = idx % F_;
  int tc = (idx / F_) % TC_;
  int c  = (idx / (F_ * TC_)) % C_;
  int b  = idx / (F_ * TC_ * C_);
  const float* src; int t0; float sgn;
  if (tc < TP_) { src = pos; t0 = tc * 5;          sgn =  1.f; }
  else          { src = neg; t0 = (tc - TP_) * 5;  sgn = -1.f; }
  const float* p = src + (((size_t)b * C_ + c) * T_ + t0) * F_ + f;
  float s = 0.f;
#pragma unroll
  for (int k = 0; k < 10; ++k) s += p[(size_t)k * F_];
  cond[(((size_t)b * C_ + c) * TC_ + tc) * F_ + f] = sgn * s * 0.1f;
}

// ---------------- K2: 1x1 conv over C + PReLU + LN(Co,F), emit bf16 --------
// One workgroup per (h,b,t). Non-transposed out: row-major [row t][Dpad].
// Transposed out (V): out[d][TCPAD_] column layout for WMMA B-operand.
__global__ __launch_bounds__(256)
void head_branch_kernel(const float* __restrict__ x, int Tx,
                        const float* __restrict__ W, const float* __restrict__ bv,
                        const float* __restrict__ a, const float* __restrict__ g,
                        const float* __restrict__ bet,
                        __bf16* __restrict__ out,
                        int Co, int Trows, int Tvalid, int Dpad, int transposed) {
  __shared__ float xs[C_ * F_];       // 33.3 KB
  __shared__ float ys[DHF_];          // 8.3 KB (max Co*F)
  __shared__ float sred[512];
  int wg  = blockIdx.x;
  int t   = wg % Trows;
  int b   = (wg / Trows) % B_;
  int h   = wg / (Trows * B_);
  int tid = threadIdx.x;
  int CoF = Co * F_;

  size_t rbase = ((size_t)(h * B_ + b) * Trows + t) * Dpad;          // row-major
  size_t tbase = (size_t)(h * B_ + b) * (size_t)CoF * TCPAD_;        // transposed

  if (t >= Tvalid) {                   // zero-fill pad rows / pad columns
    if (!transposed) { for (int o = tid; o < Dpad; o += 256) out[rbase + o] = (__bf16)0.f; }
    else             { for (int o = tid; o < CoF;  o += 256) out[tbase + (size_t)o * TCPAD_ + t] = (__bf16)0.f; }
    return;
  }

  // stage x[b,:,t,:] into LDS
  size_t xbase = ((size_t)b * C_) * Tx * F_ + (size_t)t * F_;
  for (int i = tid; i < C_ * F_; i += 256) {
    int c = i / F_, f = i % F_;
    xs[i] = x[xbase + (size_t)c * Tx * F_ + f];
  }
  __syncthreads();

  float slope = a[h];
  float s0 = 0.f, s1 = 0.f;
  for (int o = tid; o < CoF; o += 256) {
    int co = o / F_, f = o % F_;
    const float* Wr = W + (size_t)(h * Co + co) * C_;
    float acc = bv[h * Co + co];
#pragma unroll 8
    for (int c = 0; c < C_; ++c) acc += Wr[c] * xs[c * F_ + f];
    acc = (acc >= 0.f) ? acc : slope * acc;       // PReLU
    ys[o] = acc;
    s0 += acc; s1 += acc * acc;
  }
  sred[tid] = s0; sred[256 + tid] = s1;
  __syncthreads();
  for (int o = 128; o > 0; o >>= 1) {
    if (tid < o) { sred[tid] += sred[tid + o]; sred[256 + tid] += sred[256 + tid + o]; }
    __syncthreads();
  }
  float inv_n = 1.f / (float)CoF;
  float mu  = sred[0] * inv_n;
  float var = sred[256] * inv_n - mu * mu;
  float rs  = rsqrtf(var + EPS_);
  __syncthreads();

  if (!transposed) {
    for (int o = tid; o < Dpad; o += 256) {
      float v = 0.f;
      if (o < CoF) {
        int co = o / F_, f = o % F_;
        v = (ys[o] - mu) * rs * g[(size_t)(h * Co + co) * F_ + f]
            + bet[(size_t)(h * Co + co) * F_ + f];
      }
      out[rbase + o] = (__bf16)v;
    }
  } else {
    for (int o = tid; o < CoF; o += 256) {
      int co = o / F_, f = o % F_;
      float v = (ys[o] - mu) * rs * g[(size_t)(h * Co + co) * F_ + f]
                + bet[(size_t)(h * Co + co) * F_ + f];
      out[tbase + (size_t)o * TCPAD_ + t] = (__bf16)v;
    }
  }
}

// ---------------- K3: S = scale * Qf @ Kf^T  (bf16 WMMA, f32 acc) ----------
__global__ __launch_bounds__(32)
void qk_gemm_kernel(const __bf16* __restrict__ Q, const __bf16* __restrict__ K,
                    float* __restrict__ S) {
  int bid = blockIdx.x;
  int nt  = bid % NTS_;
  int mt  = (bid / NTS_) % MT_;
  int hb  = bid / (NTS_ * MT_);
  int lane = threadIdx.x;
  int mr = lane & 15, gr = lane >> 4;

  const __bf16* Qb = Q + (size_t)hb * TPAD_ * DPAD_ + (size_t)(mt * 16 + mr) * DPAD_;
  const __bf16* Kb = K + (size_t)hb * TCPAD_ * DPAD_ + (size_t)(nt * 16 + mr) * DPAD_;

  v8f acc = {};
  for (int kt = 0; kt < KTS_; ++kt) {
    if (kt + 1 < KTS_) __builtin_prefetch(Qb + (kt + 1) * 32, 0, 0);
    union { v16bf v; v8bf h[2]; } A;
    A.h[0] = *(const v8bf*)(Qb + kt * 32 + gr * 16);
    A.h[1] = *(const v8bf*)(Qb + kt * 32 + gr * 16 + 8);
    v16bf Bm = *(const v16bf*)(Kb + kt * 32 + gr * 16);
    acc = __builtin_amdgcn_wmma_f32_16x16x32_bf16(false, A.v, false, Bm,
                                                  (short)0, acc, false, false);
  }
  const float scale = 0.0438529009653515f;   // 1/sqrt(520)
  float* Sb = S + (size_t)hb * TPAD_ * TCPAD_ + (size_t)(mt * 16) * TCPAD_ + nt * 16 + mr;
#pragma unroll
  for (int r = 0; r < 8; ++r) {
    int m = gr * 8 + r;                      // D: m = (lane>>4)*8 + vgpr, n = lane&15
    Sb[(size_t)m * TCPAD_] = acc[r] * scale;
  }
}

// ---------------- K4: row softmax over s<398, emit bf16 P (zero pads) ------
__global__ __launch_bounds__(128)
void softmax_kernel(const float* __restrict__ S, __bf16* __restrict__ P) {
  __shared__ float red[128];
  int row = blockIdx.x;                      // hb*TPAD_ + t
  int t   = row % TPAD_;
  int tid = threadIdx.x;
  const float* Sr = S + (size_t)row * TCPAD_;
  __bf16* Pr = P + (size_t)row * TCPAD_;
  if (t >= T_) {
    for (int s = tid; s < TCPAD_; s += 128) Pr[s] = (__bf16)0.f;
    return;
  }
  float mx = -3.4e38f;
  for (int s = tid; s < TC_; s += 128) mx = fmaxf(mx, Sr[s]);
  red[tid] = mx; __syncthreads();
  for (int o = 64; o > 0; o >>= 1) { if (tid < o) red[tid] = fmaxf(red[tid], red[tid + o]); __syncthreads(); }
  mx = red[0]; __syncthreads();
  float sum = 0.f;
  for (int s = tid; s < TC_; s += 128) sum += __expf(Sr[s] - mx);
  red[tid] = sum; __syncthreads();
  for (int o = 64; o > 0; o >>= 1) { if (tid < o) red[tid] += red[tid + o]; __syncthreads(); }
  float inv = 1.f / red[0];
  for (int s = tid; s < TCPAD_; s += 128)
    Pr[s] = (s < TC_) ? (__bf16)(__expf(Sr[s] - mx) * inv) : (__bf16)0.f;
}

// ---------------- K5: O = P @ Vf, scatter into [B,C,T,F] f32 ---------------
__global__ __launch_bounds__(32)
void av_gemm_kernel(const __bf16* __restrict__ P, const __bf16* __restrict__ Vt,
                    float* __restrict__ O) {
  int bid = blockIdx.x;
  int nt  = bid % NTO_;
  int mt  = (bid / NTO_) % MT_;
  int hb  = bid / (NTO_ * MT_);
  int h = hb / B_, b = hb % B_;
  int lane = threadIdx.x;
  int mr = lane & 15, gr = lane >> 4;

  const __bf16* Pb = P  + (size_t)hb * TPAD_ * TCPAD_ + (size_t)(mt * 16 + mr) * TCPAD_;
  const __bf16* Vb = Vt + (size_t)hb * DHF_ * TCPAD_  + (size_t)(nt * 16 + mr) * TCPAD_;

  v8f acc = {};
  for (int kt = 0; kt < KTO_; ++kt) {
    if (kt + 1 < KTO_) __builtin_prefetch(Vb + (kt + 1) * 32, 0, 0);
    union { v16bf v; v8bf h[2]; } A;
    A.h[0] = *(const v8bf*)(Pb + kt * 32 + gr * 16);
    A.h[1] = *(const v8bf*)(Pb + kt * 32 + gr * 16 + 8);
    v16bf Bm = *(const v16bf*)(Vb + kt * 32 + gr * 16);
    acc = __builtin_amdgcn_wmma_f32_16x16x32_bf16(false, A.v, false, Bm,
                                                  (short)0, acc, false, false);
  }
  int e  = nt * 16 + mr;                     // column = dh*F + f
  int dh = e / F_, f = e % F_;
  int cc = h * DH_ + dh;
#pragma unroll
  for (int r = 0; r < 8; ++r) {
    int t = mt * 16 + gr * 8 + r;
    if (t < T_)
      O[(((size_t)b * C_ + cc) * T_ + t) * F_ + f] = acc[r];
  }
}

// ---------------- K0: Wp f32 -> bf16 ---------------------------------------
__global__ __launch_bounds__(256)
void cvt_wp_kernel(const float* __restrict__ Wp, __bf16* __restrict__ Wpb) {
  int i = blockIdx.x * 256 + threadIdx.x;
  if (i < C_ * C_) Wpb[i] = (__bf16)Wp[i];
}

// ---------------- K6: projection GEMM + PReLU + LN(C,F) per (b,t) ----------
// 8 waves: wave w owns e-rows [16w,16w+16), 5 N-tiles cover f padded to 80.
__global__ __launch_bounds__(256)
void proj_kernel(const float* __restrict__ O, const __bf16* __restrict__ Wpb,
                 const float* __restrict__ bp, const float* __restrict__ ap,
                 const float* __restrict__ gp, const float* __restrict__ betp,
                 float* __restrict__ out) {
  __shared__ __align__(32) __bf16 Xt[80 * C_];   // [f][c] 20.5 KB
  __shared__ float yb[C_ * 80];                  // [e][f] 40 KB
  __shared__ float sred[512];
  int wg  = blockIdx.x;
  int t   = wg % T_;
  int b   = wg / T_;
  int tid = threadIdx.x;

  for (int i = tid; i < 80 * C_; i += 256) {     // load X transposed, pad f>=65
    int c = i & 127, f = i >> 7;
    float v = (f < F_) ? O[(((size_t)b * C_ + c) * T_ + t) * F_ + f] : 0.f;
    Xt[f * C_ + c] = (__bf16)v;
  }
  __syncthreads();

  int wave = tid >> 5;
  int lane = tid & 31;
  int mr = lane & 15, gr = lane >> 4;
  v8f acc[5] = {};
  for (int kt = 0; kt < 4; ++kt) {
    union { v16bf v; v8bf h[2]; } A;
    const __bf16* Wr = Wpb + (size_t)(wave * 16 + mr) * C_ + kt * 32;
    A.h[0] = *(const v8bf*)(Wr + gr * 16);
    A.h[1] = *(const v8bf*)(Wr + gr * 16 + 8);
#pragma unroll
    for (int n = 0; n < 5; ++n) {
      v16bf Bm = *(const v16bf*)(&Xt[(n * 16 + mr) * C_ + kt * 32 + gr * 16]);
      acc[n] = __builtin_amdgcn_wmma_f32_16x16x32_bf16(false, A.v, false, Bm,
                                                       (short)0, acc[n], false, false);
    }
  }
  float slope = ap[0];
#pragma unroll
  for (int n = 0; n < 5; ++n) {
    int f = n * 16 + mr;
#pragma unroll
    for (int r = 0; r < 8; ++r) {
      int e = wave * 16 + gr * 8 + r;
      float y = acc[n][r] + bp[e];
      y = (y >= 0.f) ? y : slope * y;
      yb[e * 80 + f] = y;
    }
  }
  __syncthreads();

  float s0 = 0.f, s1 = 0.f;
  for (int i = tid; i < C_ * 80; i += 256) {
    int f = i % 80;
    if (f < F_) { float y = yb[i]; s0 += y; s1 += y * y; }
  }
  sred[tid] = s0; sred[256 + tid] = s1;
  __syncthreads();
  for (int o = 128; o > 0; o >>= 1) {
    if (tid < o) { sred[tid] += sred[tid + o]; sred[256 + tid] += sred[256 + tid + o]; }
    __syncthreads();
  }
  const float inv_n = 1.f / (float)(C_ * F_);
  float mu  = sred[0] * inv_n;
  float var = sred[256] * inv_n - mu * mu;
  float rs  = rsqrtf(var + EPS_);
  for (int i = tid; i < C_ * F_; i += 256) {
    int e = i / F_, f = i % F_;
    float y = (yb[e * 80 + f] - mu) * rs * gp[(size_t)e * F_ + f] + betp[(size_t)e * F_ + f];
    out[(((size_t)b * C_ + e) * T_ + t) * F_ + f] = y;
  }
}

// ---------------- host-side launch -----------------------------------------
extern "C" void kernel_launch(void* const* d_in, const int* in_sizes, int n_in,
                              void* d_out, int out_size, void* d_ws, size_t ws_size,
                              hipStream_t stream) {
  const float* batch = (const float*)d_in[0];
  const float* pos   = (const float*)d_in[1];
  const float* neg   = (const float*)d_in[2];
  const float* WQ  = (const float*)d_in[3];
  const float* bQ  = (const float*)d_in[4];
  const float* aQ  = (const float*)d_in[5];
  const float* gQ  = (const float*)d_in[6];
  const float* btQ = (const float*)d_in[7];
  const float* WK  = (const float*)d_in[8];
  const float* bK  = (const float*)d_in[9];
  const float* aK  = (const float*)d_in[10];
  const float* gK  = (const float*)d_in[11];
  const float* btK = (const float*)d_in[12];
  const float* WV  = (const float*)d_in[13];
  const float* bV  = (const float*)d_in[14];
  const float* aV  = (const float*)d_in[15];
  const float* gV  = (const float*)d_in[16];
  const float* btV = (const float*)d_in[17];
  const float* Wp  = (const float*)d_in[18];
  const float* bp  = (const float*)d_in[19];
  const float* ap  = (const float*)d_in[20];
  const float* gp  = (const float*)d_in[21];
  const float* btp = (const float*)d_in[22];
  float* out = (float*)d_out;

  char* ws = (char*)d_ws;
  size_t off = 0;
  auto carve = [&](size_t bytes) -> char* {
    char* p = ws + off;
    off = (off + bytes + 255) & ~(size_t)255;
    return p;
  };
  float*  cond = (float*) carve(sizeof(float)  * (size_t)B_ * C_ * TC_ * F_);
  __bf16* Qbf  = (__bf16*)carve(sizeof(__bf16) * 16ull * TPAD_ * DPAD_);
  __bf16* Kbf  = (__bf16*)carve(sizeof(__bf16) * 16ull * TCPAD_ * DPAD_);
  __bf16* Vtb  = (__bf16*)carve(sizeof(__bf16) * 16ull * DHF_ * TCPAD_);
  float*  S    = (float*) carve(sizeof(float)  * 16ull * TPAD_ * TCPAD_);
  __bf16* Pbf  = (__bf16*)carve(sizeof(__bf16) * 16ull * TPAD_ * TCPAD_);
  float*  O    = (float*) carve(sizeof(float)  * (size_t)B_ * C_ * T_ * F_);
  __bf16* Wpb  = (__bf16*)carve(sizeof(__bf16) * (size_t)C_ * C_);

  int tot1 = B_ * C_ * TC_ * F_;
  pool_kernel<<<(tot1 + 255) / 256, 256, 0, stream>>>(pos, neg, cond);

  head_branch_kernel<<<H_ * B_ * TPAD_, 256, 0, stream>>>(
      batch, T_, WQ, bQ, aQ, gQ, btQ, Qbf, E_, TPAD_, T_, DPAD_, 0);
  head_branch_kernel<<<H_ * B_ * TCPAD_, 256, 0, stream>>>(
      cond, TC_, WK, bK, aK, gK, btK, Kbf, E_, TCPAD_, TC_, DPAD_, 0);
  head_branch_kernel<<<H_ * B_ * TCPAD_, 256, 0, stream>>>(
      cond, TC_, WV, bV, aV, gV, btV, Vtb, DH_, TCPAD_, TC_, TCPAD_, 1);

  qk_gemm_kernel<<<16 * MT_ * NTS_, 32, 0, stream>>>(Qbf, Kbf, S);
  softmax_kernel<<<16 * TPAD_, 128, 0, stream>>>(S, Pbf);
  av_gemm_kernel<<<16 * MT_ * NTO_, 32, 0, stream>>>(Pbf, Vtb, O);

  cvt_wp_kernel<<<(C_ * C_ + 255) / 256, 256, 0, stream>>>(Wp, Wpb);
  proj_kernel<<<B_ * T_, 256, 0, stream>>>(O, Wpb, bp, ap, gp, btp, out);
}
